// FiBiNet_35527969472554
// MI455X (gfx1250) — compile-verified, matrix operand order checked by reading, not run on hardware
//
#include <hip/hip_runtime.h>
#include <hip/hip_bf16.h>

// ---------------------------------------------------------------------------
// FiBiNet forward for MI455X (gfx1250, wave32, WMMA).
//
// Sizes: B=1024, F=32, E=64, P=496, d_in=63488, h1=1024, h2=512.
// Heavy GEMMs run on v_wmma_f32_16x16x32_bf16 with f32 accumulation.
// fc GEMMs use a 64x32 wave tile (8 WMMAs / K-step, 2 weight fragments) so
// the in-register f32->bf16 weight conversion stays off the critical XDL path.
// ---------------------------------------------------------------------------

typedef __bf16 bf16;
typedef __attribute__((ext_vector_type(16))) __bf16 v16bf;
typedef __attribute__((ext_vector_type(8)))  __bf16 v8bf;
typedef __attribute__((ext_vector_type(2)))  __bf16 v2bf;
typedef __attribute__((ext_vector_type(8)))  float  v8f;

#define NUM_F   32
#define EMB     64
#define BATCH   1024
#define NPAIR   496
#define FE      2048      // F*E
#define TROW    1984      // (F-1)*E, row stride of t buffers
#define DIN     63488     // 2*P*E
#define H1      1024
#define H2      512
#define SE_RED  4

// ---------------------------------------------------------------------------
// WMMA fragment loaders.
// 16-bit A-matrix 16x32 layout (ISA 7.12.2): lane l<16 holds row M=l with
// K in {kb..kb+7, kb+16..kb+23} where kb = (l>=16)*8.  B fragment mirrors the
// layout with N across lanes; our "B" operands are stored [N,K] row-major
// (torch Linear convention y = x @ W.T), so both fragments load two
// contiguous 16-byte K-runs per lane.
// ---------------------------------------------------------------------------
__device__ __forceinline__ v16bf ld_frag_bf16(const bf16* p, long ld,
                                              int row0, int k0, int lane) {
  const int r  = lane & 15;
  const int kb = (lane >> 4) << 3;
  const bf16* q = p + (long)(row0 + r) * ld + k0 + kb;
  v8bf lo = *(const v8bf*)(q);
  v8bf hi = *(const v8bf*)(q + 16);
  return __builtin_shufflevector(lo, hi, 0, 1, 2, 3, 4, 5, 6, 7,
                                         8, 9, 10, 11, 12, 13, 14, 15);
}

// Same layout but source matrix is f32: convert to bf16 in-register
// (v_cvt_pk_bf16_f32 co-executes with XDL WMMA).
__device__ __forceinline__ v16bf ld_frag_f32(const float* p, long ld,
                                             int row0, int k0, int lane) {
  const int r  = lane & 15;
  const int kb = (lane >> 4) << 3;
  const float* q = p + (long)(row0 + r) * ld + k0 + kb;
  v8f lo = *(const v8f*)(q);
  v8f hi = *(const v8f*)(q + 16);
  v16bf out;
#pragma unroll
  for (int i = 0; i < 8; ++i) {
    out[i]     = (bf16)lo[i];
    out[i + 8] = (bf16)hi[i];
  }
  return out;
}

__device__ __forceinline__ v8f wmma_bf16(v16bf a, v16bf b, v8f c) {
  return __builtin_amdgcn_wmma_f32_16x16x32_bf16(
      /*neg_a=*/false, a, /*neg_b=*/false, b,
      /*c_mod=*/(short)0, c, /*reuse_a=*/false, /*reuse_b=*/false);
}

// ---------------------------------------------------------------------------
// Generic GEMM core: C[m,n] = act( sum_k A[m,k] * B[n,k] + bias[n] ).
// 8 waves per block; each wave owns a (TM*16 x TN*16) tile of TM*TN WMMA
// accumulators.  Waves arranged WR x WC; block tile = WR*TM*16 x WC*TN*16.
// All problem dims are multiples of the tile sizes.  ACT: 0 = none, 1 = relu.
// ---------------------------------------------------------------------------
template <int TM, int TN, int WR, int WC, int ACT, bool B_IS_F32>
__device__ __forceinline__ void gemm_core(
    const bf16* __restrict__ A, long lda,
    const void* __restrict__ Bmat, long ldb,
    const float* __restrict__ bias,
    float* __restrict__ Cf, long ldcf,
    bf16* __restrict__ Cb, long ldcb,
    int K, int m0, int n0) {
  const int lane = threadIdx.x & 31;
  const int w    = threadIdx.x >> 5;
  const int mw   = m0 + (w % WR) * (TM * 16);
  const int nw   = n0 + (w / WR) * (TN * 16);

  v8f acc[TM][TN] = {};

  for (int k = 0; k < K; k += 32) {
    if constexpr (B_IS_F32) {
      // Prefetch the streaming weight operand ~8 K-steps ahead
      // (emits global_prefetch_b8; speculative, safe at the tail).
      if (k + 256 < K) {
        __builtin_prefetch(
            (const float*)Bmat + (long)(nw + (lane & 15)) * ldb + k + 256, 0, 0);
      }
    }
    v16bf a[TM], bb[TN];
#pragma unroll
    for (int im = 0; im < TM; ++im)
      a[im] = ld_frag_bf16(A, lda, mw + im * 16, k, lane);
#pragma unroll
    for (int in = 0; in < TN; ++in) {
      if constexpr (B_IS_F32)
        bb[in] = ld_frag_f32((const float*)Bmat, ldb, nw + in * 16, k, lane);
      else
        bb[in] = ld_frag_bf16((const bf16*)Bmat, ldb, nw + in * 16, k, lane);
    }
#pragma unroll
    for (int im = 0; im < TM; ++im)
#pragma unroll
      for (int in = 0; in < TN; ++in)
        acc[im][in] = wmma_bf16(a[im], bb[in], acc[im][in]);
  }

  // C/D layout: lane l<16 -> N=l, rows M=v..v+7; lane l>=16 -> N=l-16, M=v+8.
  const int cn  = lane & 15;
  const int rhi = (lane >> 4) << 3;
#pragma unroll
  for (int im = 0; im < TM; ++im) {
#pragma unroll
    for (int in = 0; in < TN; ++in) {
      const int col = nw + in * 16 + cn;
      const float bv = bias ? bias[col] : 0.0f;
      const int rbase = mw + im * 16 + rhi;
#pragma unroll
      for (int v = 0; v < 8; ++v) {
        float val = acc[im][in][v] + bv;
        if (ACT == 1) val = fmaxf(val, 0.0f);
        const long idx_r = (long)(rbase + v);
        if (Cf) Cf[idx_r * ldcf + col] = val;
        if (Cb) Cb[idx_r * ldcb + col] = (bf16)val;
      }
    }
  }
}

// ---------------------------------------------------------------------------
// Kernel 1: SE block + f32->bf16 conversion of x and x_se.
// One block per batch row.
// ---------------------------------------------------------------------------
__global__ __launch_bounds__(256) void se_kernel(
    const float* __restrict__ x, const float* __restrict__ w1,
    const float* __restrict__ w2, bf16* __restrict__ xb,
    bf16* __restrict__ xs) {
  const int b = blockIdx.x;
  const float* xr = x + (long)b * FE;
  __shared__ float Z[NUM_F];
  __shared__ float Ah[SE_RED];
  __shared__ float G[NUM_F];
  const int t = threadIdx.x;

  if (t < NUM_F) {
    float s = 0.0f;
#pragma unroll
    for (int e = 0; e < EMB; ++e) s += xr[t * EMB + e];
    Z[t] = s * (1.0f / EMB);
  }
  __syncthreads();
  if (t < SE_RED) {
    float s = 0.0f;
#pragma unroll
    for (int f = 0; f < NUM_F; ++f) s += Z[f] * w1[t * NUM_F + f];
    Ah[t] = fmaxf(s, 0.0f);
  }
  __syncthreads();
  if (t < NUM_F) {
    float s = 0.0f;
#pragma unroll
    for (int r = 0; r < SE_RED; ++r) s += Ah[r] * w2[t * SE_RED + r];
    G[t] = 1.0f / (1.0f + __expf(-s));
  }
  __syncthreads();
  for (int idx = t; idx < FE; idx += 256) {
    const float v = xr[idx];
    const int f = idx >> 6;
    xb[(long)b * FE + idx] = (bf16)v;
    xs[(long)b * FE + idx] = (bf16)(v * G[f]);
  }
}

// ---------------------------------------------------------------------------
// Kernel 2: bilinear transforms t[b,i,:] = x[b,i,:] @ W[i].T for i in 0..30,
// for both branches.  blockIdx.z in [0,62): z<31 -> branch 1, else branch 2.
// M=1024, N=64, K=64; block tile 128x64 (2x2 wave tiles, 4x2 waves).
// ---------------------------------------------------------------------------
__global__ __launch_bounds__(256) void bilinear_kernel(
    const bf16* __restrict__ xb, const bf16* __restrict__ xs,
    const float* __restrict__ W1, const float* __restrict__ W2,
    float* __restrict__ t1, float* __restrict__ t2) {
  const int z = blockIdx.z;
  const int which = (z >= NUM_F - 1) ? 1 : 0;
  const int field = which ? (z - (NUM_F - 1)) : z;
  const bf16* A = (which ? xs : xb) + field * EMB;              // lda = FE
  const float* W = (which ? W2 : W1) + (long)field * EMB * EMB; // ldb = EMB
  float* C = (which ? t2 : t1) + field * EMB;                   // ldc = TROW
  gemm_core<2, 2, 4, 2, 0, true>(A, FE, W, EMB, nullptr,
                                 C, TROW, nullptr, 0,
                                 /*K=*/EMB, blockIdx.x * 128, 0);
}

// ---------------------------------------------------------------------------
// Kernel 3: pair products -> feat (bf16), layout [f1(P*E) | f2(P*E)] per row.
// One block per batch row: stage t1/t2/x/x_se rows in LDS (~33KB of the
// 320KB/WGP), precompute the (i,j) pair table, then stream feat out with
// packed 2xbf16 stores.
// ---------------------------------------------------------------------------
__global__ __launch_bounds__(256) void pair_kernel(
    const float* __restrict__ x, const bf16* __restrict__ xs,
    const float* __restrict__ t1, const float* __restrict__ t2,
    bf16* __restrict__ feat) {
  const int b = blockIdx.x;
  const int t = threadIdx.x;
  __shared__ float sT1[TROW];
  __shared__ float sT2[TROW];
  __shared__ float sX[FE];
  __shared__ float sXS[FE];
  __shared__ unsigned char sI[NPAIR];
  __shared__ unsigned char sJ[NPAIR];

  for (int idx = t; idx < TROW; idx += 256) {
    sT1[idx] = t1[(long)b * TROW + idx];
    sT2[idx] = t2[(long)b * TROW + idx];
  }
  for (int idx = t; idx < FE; idx += 256) {
    sX[idx]  = x[(long)b * FE + idx];
    sXS[idx] = (float)xs[(long)b * FE + idx];
  }
  for (int p = t; p < NPAIR; p += 256) {
    int i = 0, rem = p;
    while (rem >= (NUM_F - 1) - i) { rem -= (NUM_F - 1) - i; ++i; }
    sI[p] = (unsigned char)i;
    sJ[p] = (unsigned char)(i + 1 + rem);
  }
  __syncthreads();

  // 256 threads handle 8 pairs x 32 element-pairs per iteration.
  const int e    = (t & 31) * 2;
  const int psub = t >> 5;
  bf16* fr1 = feat + (long)b * DIN;
  bf16* fr2 = fr1 + NPAIR * EMB;
  for (int p0 = 0; p0 < NPAIR; p0 += 8) {
    const int p = p0 + psub;
    const int i = sI[p], j = sJ[p];
    const int ti = i * EMB + e, xj = j * EMB + e;
    const float a0 = sT1[ti]     * sX[xj];
    const float a1 = sT1[ti + 1] * sX[xj + 1];
    const float c0 = sT2[ti]     * sXS[xj];
    const float c1 = sT2[ti + 1] * sXS[xj + 1];
    v2bf o1 = {(bf16)a0, (bf16)a1};
    v2bf o2 = {(bf16)c0, (bf16)c1};
    *(v2bf*)(fr1 + p * EMB + e) = o1;
    *(v2bf*)(fr2 + p * EMB + e) = o2;
  }
}

// ---------------------------------------------------------------------------
// Kernel 4/5: fused GEMM + bias + relu, bf16 activations, f32 weights
// converted in-register.  64x32 wave tile, block tile 128x128 (2x4 waves):
// 8 WMMAs per K-step with only 2 weight fragments.
// ---------------------------------------------------------------------------
template <int ACT>
__global__ __launch_bounds__(256) void fc_gemm_kernel(
    const bf16* __restrict__ A, long lda,
    const float* __restrict__ W, long ldb,
    const float* __restrict__ bias,
    bf16* __restrict__ Cb, long ldcb, int K) {
  gemm_core<4, 2, 2, 4, ACT, true>(A, lda, W, ldb, bias,
                                   nullptr, 0, Cb, ldcb,
                                   K, blockIdx.x * 128, blockIdx.y * 128);
}

// ---------------------------------------------------------------------------
// Kernel 6: fc2 (512 -> 1) + sigmoid.  One wave per batch row.
// ---------------------------------------------------------------------------
__global__ __launch_bounds__(32) void fc2_kernel(
    const bf16* __restrict__ h, const float* __restrict__ w,
    const float* __restrict__ bias, float* __restrict__ out) {
  const int b = blockIdx.x;
  const int t = threadIdx.x;
  float s = 0.0f;
#pragma unroll 4
  for (int k = t; k < H2; k += 32) s += (float)h[(long)b * H2 + k] * w[k];
#pragma unroll
  for (int off = 16; off > 0; off >>= 1) s += __shfl_xor(s, off, 32);
  if (t == 0) out[b] = 1.0f / (1.0f + __expf(-(s + bias[0])));
}

// ---------------------------------------------------------------------------
// Launcher
// ---------------------------------------------------------------------------
extern "C" void kernel_launch(void* const* d_in, const int* in_sizes, int n_in,
                              void* d_out, int out_size, void* d_ws,
                              size_t ws_size, hipStream_t stream) {
  (void)in_sizes; (void)n_in; (void)out_size; (void)ws_size;

  const float* x     = (const float*)d_in[0];   // (B, F, E)
  const float* W1    = (const float*)d_in[1];   // (31, E, E)
  const float* W2    = (const float*)d_in[2];   // (31, E, E)
  const float* se_w1 = (const float*)d_in[3];   // (4, 32)
  const float* se_w2 = (const float*)d_in[4];   // (32, 4)
  const float* fc0_w = (const float*)d_in[5];   // (1024, 63488)
  const float* fc0_b = (const float*)d_in[6];
  const float* fc1_w = (const float*)d_in[7];   // (512, 1024)
  const float* fc1_b = (const float*)d_in[8];
  const float* fc2_w = (const float*)d_in[9];   // (1, 512)
  const float* fc2_b = (const float*)d_in[10];
  float* out = (float*)d_out;                   // (B, 1)

  char* ws = (char*)d_ws;
  auto alloc = [&](size_t bytes) -> char* {
    char* p = ws;
    ws += (bytes + 255) & ~(size_t)255;
    return p;
  };
  bf16* xb   = (bf16*)alloc((size_t)BATCH * FE * sizeof(bf16));     //   4 MB
  bf16* xs   = (bf16*)alloc((size_t)BATCH * FE * sizeof(bf16));     //   4 MB
  float* t1  = (float*)alloc((size_t)BATCH * TROW * sizeof(float)); //   8 MB
  float* t2  = (float*)alloc((size_t)BATCH * TROW * sizeof(float)); //   8 MB
  bf16* feat = (bf16*)alloc((size_t)BATCH * DIN * sizeof(bf16));    // 130 MB
  bf16* h0b  = (bf16*)alloc((size_t)BATCH * H1 * sizeof(bf16));     //   2 MB
  bf16* h1b  = (bf16*)alloc((size_t)BATCH * H2 * sizeof(bf16));     //   1 MB

  // 1) SE gates + bf16 conversion
  se_kernel<<<BATCH, 256, 0, stream>>>(x, se_w1, se_w2, xb, xs);

  // 2) Bilinear transforms (both branches), WMMA
  bilinear_kernel<<<dim3(BATCH / 128, 1, 2 * (NUM_F - 1)), 256, 0, stream>>>(
      xb, xs, W1, W2, t1, t2);

  // 3) Pair products -> feat (bf16)
  pair_kernel<<<BATCH, 256, 0, stream>>>(x, xs, t1, t2, feat);

  // 4) fc0: (1024 x 63488) @ (63488 x 1024)^T + bias, relu  [dominant GEMM]
  fc_gemm_kernel<1><<<dim3(BATCH / 128, H1 / 128), 256, 0, stream>>>(
      feat, DIN, fc0_w, DIN, fc0_b, h0b, H1, DIN);

  // 5) fc1: (1024 x 1024) @ (1024 x 512)^T + bias, relu
  fc_gemm_kernel<1><<<dim3(BATCH / 128, H2 / 128), 256, 0, stream>>>(
      h0b, H1, fc1_w, H1, fc1_b, h1b, H2, H1);

  // 6) fc2 + sigmoid
  fc2_kernel<<<BATCH, 32, 0, stream>>>(h1b, fc2_w, fc2_b, out);
}
